// VectorQuantizer_51659866636801
// MI455X (gfx1250) — compile-verified, matrix operand order checked by reading, not run on hardware
//
#include <hip/hip_runtime.h>
#include <math.h>

// ---------------- sizes ----------------
#define DIMX   256
#define DC     1024
#define KCODE  1024
#define TTOK   65536          // 16 * 4096

// ---------------- bf16 helpers ----------------
typedef __attribute__((ext_vector_type(16))) __bf16 v16bf;
typedef __attribute__((ext_vector_type(8)))  __bf16 v8bf;
typedef __attribute__((ext_vector_type(8)))  float  v8f;

__device__ __forceinline__ unsigned short f2bf(float f) {
    union { float f; unsigned u; } v; v.f = f;
    unsigned r = v.u + 0x7FFFu + ((v.u >> 16) & 1u);   // round-to-nearest-even
    return (unsigned short)(r >> 16);
}
__device__ __forceinline__ float bf2f(unsigned short b) {
    union { unsigned u; float f; } v; v.u = ((unsigned)b) << 16;
    return v.f;
}

union V16U { v16bf v; v8bf h[2]; };

// Load a 16x32 bf16 A-frag (or 32x16 B-frag) from a row-major tile whose
// second index (K) is contiguous.  ISA layout: lanes 0-15 -> row=lane,
// K={0..7,16..23}; lanes 16-31 -> row=lane-16, K={8..15,24..31}.
__device__ __forceinline__ v16bf load_frag(const unsigned short* base, int stride, int lane) {
    int r    = lane & 15;
    int koff = (lane >> 4) << 3;                 // 0 or 8
    const unsigned short* p = base + r * stride + koff;
    V16U u;
    u.h[0] = *reinterpret_cast<const v8bf*>(p);
    u.h[1] = *reinterpret_cast<const v8bf*>(p + 16);
    return u.v;
}

#define WMMA_BF16(a, b, c) \
    __builtin_amdgcn_wmma_f32_16x16x32_bf16(false, (a), false, (b), (short)0, (c), false, false)

// ---------------- async global -> LDS (CDNA5) ----------------
// Builtin signature (from hipcc diagnostics): param1 = int4 addrspace(1)*,
// param2 = int4 addrspace(3)*, then imm offset, imm cpol.
typedef int v4i_vs __attribute__((vector_size(16)));

#if __has_builtin(__builtin_amdgcn_global_load_async_to_lds_b128)
#define HAVE_ASYNC_LDS 1
#define ASYNC_B128(gsrc, ldst)                                          \
    __builtin_amdgcn_global_load_async_to_lds_b128(                     \
        (__attribute__((address_space(1))) v4i_vs*)(gsrc),              \
        (__attribute__((address_space(3))) v4i_vs*)(ldst), 0, 0)
#if __has_builtin(__builtin_amdgcn_s_wait_asynccnt)
#define WAIT_ASYNC(n) __builtin_amdgcn_s_wait_asynccnt(n)
#else
#define WAIT_ASYNC_STR(n) asm volatile("s_wait_asynccnt " #n)
#define WAIT_ASYNC(n) WAIT_ASYNC_STR(n)
#endif
#else
#define HAVE_ASYNC_LDS 0
#define ASYNC_B128(gsrc, ldst)                                          \
    (*reinterpret_cast<uint4*>(ldst) = *reinterpret_cast<const uint4*>(gsrc))
#define WAIT_ASYNC(n)
#endif

// ---------------- prep kernels ----------------
// w1 (256 x 1024) -> w1b transposed bf16 (1024 x 256), contiguous in K
__global__ void k_prep_w1(const float* __restrict__ w1, unsigned short* __restrict__ w1b) {
    int e = blockIdx.x * blockDim.x + threadIdx.x;      // 0..262143
    int n = e >> 8, k = e & 255;
    w1b[e] = f2bf(w1[k * DC + n]);
}

// embed (1024 x 1024) -> bf16 + e_sq per row
__global__ void k_prep_embed(const float* __restrict__ embed,
                             unsigned short* __restrict__ embedb,
                             float* __restrict__ e_sq) {
    __shared__ float red[8];
    int r = blockIdx.x, tid = threadIdx.x;
    float s = 0.f;
#pragma unroll
    for (int i = 0; i < 4; ++i) {
        int c = tid + i * 256;
        float v = embed[r * DC + c];
        embedb[r * DC + c] = f2bf(v);
        s += v * v;
    }
    for (int off = 16; off > 0; off >>= 1) s += __shfl_xor(s, off, 32);
    int wave = tid >> 5, lane = tid & 31;
    if (lane == 0) red[wave] = s;
    __syncthreads();
    if (tid == 0) {
        float t = 0.f;
        for (int i = 0; i < 8; ++i) t += red[i];
        e_sq[r] = t;
    }
}

// proj = embed @ w2 + b2   (1024 x 256, tiny)
__global__ void k_prep_proj(const float* __restrict__ embed, const float* __restrict__ w2,
                            const float* __restrict__ b2, float* __restrict__ proj) {
    int r = blockIdx.x, c = threadIdx.x;
    float s = b2[c];
    const float* er = embed + r * DC;
    for (int k = 0; k < DC; ++k) s = fmaf(er[k], w2[k * DIMX + c], s);
    proj[r * DIMX + c] = s;
}

// ---------------- LayerNorm + GELU -> bf16 act ----------------
__global__ void k_ln_gelu(const float* __restrict__ x, const float* __restrict__ ln_w,
                          const float* __restrict__ ln_b, unsigned short* __restrict__ act) {
    int wave = threadIdx.x >> 5, lane = threadIdx.x & 31;
    int row  = blockIdx.x * 8 + wave;
    const float* xr = x + (size_t)row * DIMX;
    float v[8];
    float4 a = *reinterpret_cast<const float4*>(xr + lane * 8);
    float4 b = *reinterpret_cast<const float4*>(xr + lane * 8 + 4);
    v[0]=a.x; v[1]=a.y; v[2]=a.z; v[3]=a.w; v[4]=b.x; v[5]=b.y; v[6]=b.z; v[7]=b.w;
    float s = 0.f, sq = 0.f;
#pragma unroll
    for (int i = 0; i < 8; ++i) { s += v[i]; sq += v[i] * v[i]; }
    for (int off = 16; off > 0; off >>= 1) { s += __shfl_xor(s, off, 32); sq += __shfl_xor(sq, off, 32); }
    float mean = s * (1.f / DIMX);
    float var  = sq * (1.f / DIMX) - mean * mean;
    float rs   = rsqrtf(var + 1e-5f);
    unsigned short* ar = act + (size_t)row * DIMX;
#pragma unroll
    for (int i = 0; i < 8; ++i) {
        int c = lane * 8 + i;
        float xn = (v[i] - mean) * rs * ln_w[c] + ln_b[c];
        float g  = 0.5f * xn * (1.f + erff(xn * 0.70710678118654752f));
        ar[c] = f2bf(g);
    }
}

// ---------------- GEMM1: z = act @ w1 + b1  (bf16 WMMA) ----------------
// 64-row blocks, 8 waves: wave = m_strip(4) x n_half(2)
__global__ void k_gemm1(const unsigned short* __restrict__ act,
                        const unsigned short* __restrict__ w1b,
                        const float* __restrict__ b1,
                        unsigned short* __restrict__ z) {
    __shared__ alignas(16) unsigned short As[64 * DIMX];      // 32 KB
    int tid = threadIdx.x, wave = tid >> 5, lane = tid & 31;
    int m_strip = wave >> 1, n_half = wave & 1;
    int t0 = blockIdx.x * 64;

    // stage A: 64x256 bf16, async straight into LDS
#pragma unroll
    for (int i = 0; i < 8; ++i) {
        int c   = tid + i * 256;                              // 2048 chunks of 8
        int row = c >> 5, ko = (c & 31) << 3;
        ASYNC_B128(act + (size_t)(t0 + row) * DIMX + ko, As + row * DIMX + ko);
    }
    WAIT_ASYNC(0);
    __syncthreads();

    const unsigned short* Abase = As + m_strip * 16 * DIMX;
    for (int ntl = 0; ntl < 32; ++ntl) {
        int nt = n_half * 32 + ntl;
        v8f acc = {};
#pragma unroll
        for (int kt = 0; kt < 8; ++kt) {
            v16bf a = load_frag(Abase + kt * 32, DIMX, lane);
            v16bf b = load_frag(w1b + (size_t)nt * 16 * DIMX + kt * 32, DIMX, lane);
            acc = WMMA_BF16(a, b, acc);
        }
        int colg  = nt * 16 + (lane & 15);
        float bias = b1[colg];
        int rbase = t0 + m_strip * 16 + ((lane >> 4) << 3);
#pragma unroll
        for (int r = 0; r < 8; ++r)
            z[(size_t)(rbase + r) * DC + colg] = f2bf(acc[r] + bias);
    }
}

// ---------------- z_sq per token ----------------
__global__ void k_zsq(const unsigned short* __restrict__ z, float* __restrict__ zsq) {
    int wave = threadIdx.x >> 5, lane = threadIdx.x & 31;
    int row  = blockIdx.x * 8 + wave;
    const unsigned short* zr = z + (size_t)row * DC;
    float s = 0.f;
#pragma unroll
    for (int i = 0; i < 4; ++i) {
        int c = lane * 32 + i * 8;
        uint4 pk = *reinterpret_cast<const uint4*>(zr + c);
        const unsigned short* u = reinterpret_cast<const unsigned short*>(&pk);
#pragma unroll
        for (int j = 0; j < 8; ++j) { float v = bf2f(u[j]); s += v * v; }
    }
    for (int off = 16; off > 0; off >>= 1) s += __shfl_xor(s, off, 32);
    if (lane == 0) zsq[row] = s;
}

__global__ void k_init_loss(float* loss) { *loss = 0.f; }

// ---------------- distance GEMM + argmin + loss ----------------
// 128-row blocks, 8 waves each owning a 16-row strip; N processed in 8 chunks
// of 128 codes; K in 32 tiles of 32.  Double-buffered async LDS staging:
// issue tile kt+1 while WMMAs consume tile kt.
__global__ void k_dist_argmin(const unsigned short* __restrict__ z,
                              const unsigned short* __restrict__ embedb,
                              const float* __restrict__ e_sq,
                              const float* __restrict__ zsq,
                              int* __restrict__ idx_ws,
                              float* __restrict__ idx_out,
                              float* __restrict__ loss) {
    __shared__ alignas(16) unsigned short As[2][128 * 32];    // 2 x 8 KB
    __shared__ alignas(16) unsigned short Bs[2][128 * 32];    // 2 x 8 KB
    int tid = threadIdx.x, wave = tid >> 5, lane = tid & 31;
    int t0 = blockIdx.x * 128;

    float best[8]; int bidx[8];
#pragma unroll
    for (int r = 0; r < 8; ++r) { best[r] = 3.0e38f; bidx[r] = 0; }

    for (int nc = 0; nc < 8; ++nc) {
        if (nc + 1 < 8)   // warm L2/WGP$ for next codebook chunk
            __builtin_prefetch(embedb + (size_t)((nc + 1) * 128 + (tid >> 1)) * DC, 0, 2);

        v8f acc[8];
#pragma unroll
        for (int nt = 0; nt < 8; ++nt) acc[nt] = (v8f){};

        // prologue: stage K-tile 0 into buffer 0 (4 async ops per wave)
#pragma unroll
        for (int i = 0; i < 2; ++i) {
            int c = tid + i * 256;                            // 512 chunks of 8
            int row = c >> 2, ko = (c & 3) << 3;
            ASYNC_B128(z + (size_t)(t0 + row) * DC + ko, &As[0][row * 32 + ko]);
            ASYNC_B128(embedb + (size_t)(nc * 128 + row) * DC + ko, &Bs[0][row * 32 + ko]);
        }

        for (int kt = 0; kt < 32; ++kt) {
            int cur = kt & 1;
            if (kt + 1 < 32) {                                // stage next tile
#pragma unroll
                for (int i = 0; i < 2; ++i) {
                    int c = tid + i * 256;
                    int row = c >> 2, ko = (c & 3) << 3;
                    ASYNC_B128(z + (size_t)(t0 + row) * DC + (kt + 1) * 32 + ko,
                               &As[cur ^ 1][row * 32 + ko]);
                    ASYNC_B128(embedb + (size_t)(nc * 128 + row) * DC + (kt + 1) * 32 + ko,
                               &Bs[cur ^ 1][row * 32 + ko]);
                }
                WAIT_ASYNC(4);                                // tile kt landed
            } else {
                WAIT_ASYNC(0);
            }
            __syncthreads();                                  // all waves' tile kt visible

            v16bf a = load_frag(&As[cur][wave * 16 * 32], 32, lane);
#pragma unroll
            for (int nt = 0; nt < 8; ++nt) {
                v16bf b = load_frag(&Bs[cur][nt * 16 * 32], 32, lane);
                acc[nt] = WMMA_BF16(a, b, acc[nt]);
            }
            __syncthreads();                                  // readers done before restage
        }

        // fold scores into running argmin: dist-part = e_sq - 2*s
#pragma unroll
        for (int nt = 0; nt < 8; ++nt) {
            int code = nc * 128 + nt * 16 + (lane & 15);
            float es = e_sq[code];
#pragma unroll
            for (int r = 0; r < 8; ++r) {
                float v = es - 2.0f * acc[nt][r];
                if (v < best[r]) { best[r] = v; bidx[r] = code; }
            }
        }
    }

    // cross-lane min within 16-lane groups (same row set), jnp tiebreak = lowest idx
#pragma unroll
    for (int r = 0; r < 8; ++r) {
        for (int off = 1; off < 16; off <<= 1) {
            float ov = __shfl_xor(best[r], off, 16);
            int   oi = __shfl_xor(bidx[r], off, 16);
            if (ov < best[r] || (ov == best[r] && oi < bidx[r])) { best[r] = ov; bidx[r] = oi; }
        }
    }
    if ((lane & 15) == 0) {
        int rbase = t0 + wave * 16 + ((lane >> 4) << 3);
        float lsum = 0.f;
#pragma unroll
        for (int r = 0; r < 8; ++r) {
            int row = rbase + r;
            idx_ws[row]  = bidx[r];
            idx_out[row] = (float)bidx[r];
            lsum += best[r] + zsq[row];                       // min squared distance
        }
        atomicAdd(loss, lsum * (1.25f / ((float)TTOK * (float)DC)));
    }
}

// ---------------- gather: quantized = proj[idx] ----------------
__global__ void k_gather(const int* __restrict__ idx_ws, const float* __restrict__ proj,
                         float* __restrict__ out) {
    int t = blockIdx.x, c = threadIdx.x;
    out[(size_t)t * DIMX + c] = proj[(size_t)idx_ws[t] * DIMX + c];
}

// ---------------- launch ----------------
extern "C" void kernel_launch(void* const* d_in, const int* in_sizes, int n_in,
                              void* d_out, int out_size, void* d_ws, size_t ws_size,
                              hipStream_t stream) {
    (void)in_sizes; (void)n_in; (void)out_size; (void)ws_size;
    const float* x    = (const float*)d_in[0];
    const float* ln_w = (const float*)d_in[1];
    const float* ln_b = (const float*)d_in[2];
    const float* w1   = (const float*)d_in[3];
    const float* b1   = (const float*)d_in[4];
    const float* emb  = (const float*)d_in[5];
    const float* w2   = (const float*)d_in[6];
    const float* b2   = (const float*)d_in[7];

    char* ws = (char*)d_ws;
    size_t o = 0;
    auto take = [&](size_t bytes) { size_t cur = o; o = (o + bytes + 255) & ~(size_t)255; return cur; };
    unsigned short* act    = (unsigned short*)(ws + take((size_t)TTOK * DIMX * 2));
    unsigned short* z      = (unsigned short*)(ws + take((size_t)TTOK * DC * 2));
    float*          zsq    = (float*)         (ws + take((size_t)TTOK * 4));
    unsigned short* w1b    = (unsigned short*)(ws + take((size_t)DIMX * DC * 2));
    unsigned short* embedb = (unsigned short*)(ws + take((size_t)KCODE * DC * 2));
    float*          esq    = (float*)         (ws + take((size_t)KCODE * 4));
    float*          proj   = (float*)         (ws + take((size_t)KCODE * DIMX * 4));
    int*            idxw   = (int*)           (ws + take((size_t)TTOK * 4));

    float* qout = (float*)d_out;                 // (16,4096,256)
    float* iout = qout + (size_t)TTOK * DIMX;    // (16,4096) as float
    float* loss = iout + TTOK;                   // scalar

    k_prep_w1   <<<(DIMX * DC) / 256, 256, 0, stream>>>(w1, w1b);
    k_prep_embed<<<KCODE, 256, 0, stream>>>(emb, embedb, esq);
    k_prep_proj <<<KCODE, DIMX, 0, stream>>>(emb, w2, b2, proj);
    k_ln_gelu   <<<TTOK / 8, 256, 0, stream>>>(x, ln_w, ln_b, act);
    k_gemm1     <<<TTOK / 64, 256, 0, stream>>>(act, w1b, b1, z);
    k_zsq       <<<TTOK / 8, 256, 0, stream>>>(z, zsq);
    k_init_loss <<<1, 1, 0, stream>>>(loss);
    k_dist_argmin<<<TTOK / 128, 256, 0, stream>>>(z, embedb, esq, zsq, idxw, iout, loss);
    k_gather    <<<TTOK, DIMX, 0, stream>>>(idxw, proj, qout);
}
